// MultiHeadedAttention_76819785056273
// MI455X (gfx1250) — matrix operand order repaired and checked by reading, not compiled, not run on hardware
//
#include <hip/hip_runtime.h>

// ---------------------------------------------------------------------------
// Problem constants (fixed by the reference): B=4, S=2048, H=16, D=1024, DK=64
// ---------------------------------------------------------------------------
constexpr int Bc = 4, Hc = 16, Sc = 2048, Dc = 1024, DKc = 64;
constexpr int Mrows = Bc * Sc;  // 8192 rows for the projection GEMMs

typedef __bf16 bf16_t;
typedef bf16_t v8bf  __attribute__((ext_vector_type(8)));
typedef bf16_t v16bf __attribute__((ext_vector_type(16)));
typedef float  v8f   __attribute__((ext_vector_type(8)));
typedef unsigned int u32x4 __attribute__((ext_vector_type(4)));
typedef int          i32x4 __attribute__((ext_vector_type(4)));
typedef int          i32x8 __attribute__((ext_vector_type(8)));

#define DEV_INLINE __device__ __forceinline__

// --- Tensor Data Mover availability / arity detection ----------------------
// amdgpu-toolchain (clang-23 + therock-10.0 headers) ships the TDM header and
// uses the 6-arg builtin; ROCm 7.2 (clang-22) has the 5-arg form.
#if defined(__has_include)
#  if __has_include(<hip/amd_detail/amd_gfx1250_TDM.h>)
#    define TDM_SIX_ARGS 1
#  else
#    define TDM_SIX_ARGS 0
#  endif
#else
#  define TDM_SIX_ARGS 0
#endif
#if defined(__has_builtin)
#  if __has_builtin(__builtin_amdgcn_tensor_load_to_lds)
#    define HAVE_TDM 1
#  endif
#endif
#ifndef HAVE_TDM
#  define HAVE_TDM 0
#endif

// round-to-nearest-even f32 -> bf16 (no reliance on __hip_bfloat16 helpers)
DEV_INLINE bf16_t f2bf(float f) {
    unsigned u = __builtin_bit_cast(unsigned, f);
    unsigned r = u + 0x7FFFu + ((u >> 16) & 1u);
    unsigned short h = (unsigned short)(r >> 16);
    return __builtin_bit_cast(bf16_t, h);
}

DEV_INLINE v16bf cat8(v8bf lo, v8bf hi) {
    return __builtin_shufflevector(lo, hi, 0,1,2,3,4,5,6,7,8,9,10,11,12,13,14,15);
}

// A-operand fragment of v_wmma_*_16x16x32_bf16.  Source tile is row-major
// [16 rows][>=32 cols], `stride` elements per row (16B aligned).
// ISA layout (CDNA5 ISA): lanes 0-15 hold row M=lane, K={0..7,16..23};
// lanes 16-31 hold row M=lane-16, K={8..15,24..31}.
DEV_INLINE v16bf load_fragA(const bf16_t* p, int stride, int lane) {
    const int m  = lane & 15;
    const int kb = (lane >> 4) << 3;          // 0 or 8
    const bf16_t* q = p + m * stride;
    v8bf lo = *(const v8bf*)(q + kb);         // K = kb .. kb+7
    v8bf hi = *(const v8bf*)(q + 16 + kb);    // K = 16+kb .. 23+kb
    return cat8(lo, hi);
}

// B-operand fragment.  Source stored "N-major" (B^T): row n holds the
// reduction axis contiguously.  ISA layout: lanes 0-15 -> K={0..7,16..23} of
// column N=lane, lanes 16-31 -> K={8..15,24..31} of column N=lane-16.
DEV_INLINE v16bf load_fragB(const bf16_t* p, int stride, int lane) {
    const int n  = lane & 15;
    const int kb = (lane >> 4) << 3;          // 0 or 8
    const bf16_t* q = p + n * stride + kb;
    v8bf lo = *(const v8bf*)(q);
    v8bf hi = *(const v8bf*)(q + 16);
    return cat8(lo, hi);
}

DEV_INLINE v8f wmma_bf16(v16bf a, v16bf b, v8f c) {
    // (neg_a, A, neg_b, B, c_mod, C, reuse_a, reuse_b)
    return __builtin_amdgcn_wmma_f32_16x16x32_bf16(false, a, false, b,
                                                   (short)0, c, false, false);
}

#if HAVE_TDM
// ---------------------------------------------------------------------------
// TDM 2D tile load: global (row-major, stride0 elems/row, bf16) -> LDS,
// optional LDS row padding.  D# packing per CDNA5 ISA
// §8.3/8.4: group0 = {count|flags, lds_addr, global_addr[31:0],
// global_addr[56:32]|type=2}; group1 carries data_size/pad, tensor dims,
// tile dims and the 48-bit dim0 stride.  Issued once per wave (EXEC ignored),
// completion via TENSORcnt.
// ---------------------------------------------------------------------------
DEV_INLINE void tdm_load_2d_bf16(unsigned lds_off, const bf16_t* gptr,
                                 unsigned tile_d0, unsigned tile_d1,
                                 unsigned long long stride0_elems,
                                 unsigned pad_interval, unsigned pad_amount,
                                 bool pad_en) {
    const unsigned long long ga = (unsigned long long)(uintptr_t)gptr;
    u32x4 g0;
    g0[0] = 1u;                                         // count=1, user D#
    g0[1] = lds_off;                                    // lds_addr (bytes)
    g0[2] = (unsigned)(ga & 0xFFFFFFFFu);               // global_addr lo
    g0[3] = (unsigned)((ga >> 32) & 0x01FFFFFFu) | (2u << 30);  // hi | type=2
    const unsigned tens_d0 = 0xFFFFFu, tens_d1 = 0xFFFFFu;      // no OOB clip
    unsigned d0 = (1u << 16);                           // data_size = 2 bytes
    if (pad_en) d0 |= (1u << 20) | (pad_interval << 22) | (pad_amount << 25);
    i32x8 g1;
    g1[0] = (int)d0;                                    // mask=0 (no cluster)
    g1[1] = (int)((tens_d0 & 0xFFFFu) << 16);           // dim0[15:0]@63:48
    g1[2] = (int)(((tens_d0 >> 16) & 0xFFFFu) | ((tens_d1 & 0xFFFFu) << 16));
    g1[3] = (int)(((tens_d1 >> 16) & 0xFFFFu) | ((tile_d0 & 0xFFFFu) << 16));
    g1[4] = (int)(tile_d1 & 0xFFFFu);                   // tile_dim1, dim2=0
    g1[5] = (int)(unsigned)(stride0_elems & 0xFFFFFFFFull);
    g1[6] = (int)(unsigned)((stride0_elems >> 32) & 0xFFFFull);  // stride1=0
    g1[7] = 0;
    const i32x4 z4 = {};
#if TDM_SIX_ARGS
    const i32x8 z8 = {};
    __builtin_amdgcn_tensor_load_to_lds(g0, g1, z4, z4, z8, 0);
#else
    __builtin_amdgcn_tensor_load_to_lds(g0, g1, z4, z4, 0);
#endif
}
#endif  // HAVE_TDM

// ---------------------------------------------------------------------------
// GEMM with fused bias:  C[m][n] = sum_k A[m][k] * W[n][k] + bias[n]
// Block: 256 threads = 8 waves -> 64x32 output tile, KC=64 K-panel in LDS
// (2 wmma per wave per stage).  bf16 A panels are staged by the TDM.
// OUT_MODE 0: bf16 out at [b,h,s,dk]   (Q, K)
// OUT_MODE 1: bf16 out at [b,h,dk,s]   (V transposed for the PV wmma)
// OUT_MODE 2: f32  out row-major [m][n] (final projection)
// ---------------------------------------------------------------------------
constexpr int BM = 64, BN = 32, KC = 64;
constexpr int LDA_ = KC + 8, LDB_ = KC + 8;   // 144B row stride: 16B aligned

template <int OUT_MODE, bool A_BF16>
__global__ void __launch_bounds__(256)
gemm_wmma_bias(const void* __restrict__ Av, const float* __restrict__ W,
               const float* __restrict__ bias, void* __restrict__ Out,
               int M, int N, int K) {
    __shared__ bf16_t As[BM * LDA_];
    __shared__ bf16_t Bs[BN * LDB_];
    const int tid  = threadIdx.x;
    const int lane = tid & 31;
    const int w    = tid >> 5;        // 8 waves
    const int wr   = w >> 1;          // 0..3  (16-row strip)
    const int wc   = w & 1;           // 0..1  (16-col strip)
    const int m0   = blockIdx.x * BM;
    const int n0   = blockIdx.y * BN;

    v8f acc = {};

    for (int k0 = 0; k0 < K; k0 += KC) {
        // ---- stage A panel (64 x 64) into LDS as bf16 ----
        if constexpr (A_BF16) {
            const bf16_t* A = (const bf16_t*)Av;
#if HAVE_TDM
            // One TDM descriptor per workgroup: 64x64 bf16 tile, HW-applied
            // +16B LDS row pad (row = 128B = 2^4 x 8B units -> interval 4;
            // pad 4 dwords -> amount code 3).
            if (w == 0) {
                tdm_load_2d_bf16((unsigned)(uintptr_t)&As[0],
                                 A + (size_t)m0 * K + k0,
                                 /*tile_d0=*/KC, /*tile_d1=*/BM,
                                 /*stride0=*/(unsigned long long)K,
                                 /*pad_interval=*/4u, /*pad_amount=*/3u, true);
                __builtin_amdgcn_s_wait_tensorcnt(0);
            }
#else
#pragma unroll
            for (int i = 0; i < 2; ++i) {
                const int lin = tid + i * 256;
                const int r = lin >> 3, seg = lin & 7; // 64 rows x 8 segs of 8
                *(v8bf*)&As[r * LDA_ + seg * 8] =
                    *(const v8bf*)(A + (size_t)(m0 + r) * K + k0 + seg * 8);
            }
#endif
        } else {
            const float* A = (const float*)Av;
#pragma unroll
            for (int i = 0; i < 4; ++i) {
                const int lin = tid + i * 256;
                const int r = lin >> 4, seg = lin & 15;  // 64 x 16 segs of 4
                const float4 f =
                    *(const float4*)(A + (size_t)(m0 + r) * K + k0 + seg * 4);
                bf16_t* d = &As[r * LDA_ + seg * 4];
                d[0] = f2bf(f.x); d[1] = f2bf(f.y);
                d[2] = f2bf(f.z); d[3] = f2bf(f.w);
            }
        }
        // ---- stage W panel (32 x 64), f32 -> bf16 ----
#pragma unroll
        for (int i = 0; i < 2; ++i) {
            const int lin = tid + i * 256;
            const int r = lin >> 4, seg = lin & 15;      // 32 x 16 segs of 4
            const float4 f =
                *(const float4*)(W + (size_t)(n0 + r) * K + k0 + seg * 4);
            bf16_t* d = &Bs[r * LDB_ + seg * 4];
            d[0] = f2bf(f.x); d[1] = f2bf(f.y);
            d[2] = f2bf(f.z); d[3] = f2bf(f.w);
        }
        __syncthreads();
        const bf16_t* arow = &As[(wr * 16) * LDA_];
        const bf16_t* brow = &Bs[(wc * 16) * LDB_];
        acc = wmma_bf16(load_fragA(arow, LDA_, lane),
                        load_fragB(brow, LDB_, lane), acc);
        acc = wmma_bf16(load_fragA(arow + 32, LDA_, lane),
                        load_fragB(brow + 32, LDB_, lane), acc);
        __syncthreads();
    }

    // ---- epilogue: C layout = lane(n + 16*(m>=8)), vgpr = m%8 ----
    const int mhalf = (lane >> 4) << 3;   // 0 or 8
    const int nn    = lane & 15;
    const int gn    = n0 + wc * 16 + nn;
    const float bb  = bias[gn];
#pragma unroll
    for (int vg = 0; vg < 8; ++vg) {
        const int gm    = m0 + wr * 16 + vg + mhalf;
        const float val = acc[vg] + bb;
        if constexpr (OUT_MODE == 2) {
            ((float*)Out)[(size_t)gm * N + gn] = val;
        } else {
            const int b_ = gm >> 11, s_ = gm & (Sc - 1);   // S = 2048
            const int h_ = gn >> 6,  dk = gn & (DKc - 1);  // DK = 64
            size_t idx;
            if constexpr (OUT_MODE == 0)
                idx = ((size_t)(b_ * Hc + h_) * Sc + s_) * DKc + dk;
            else
                idx = ((size_t)(b_ * Hc + h_) * DKc + dk) * Sc + s_;
            ((bf16_t*)Out)[idx] = f2bf(val);
        }
    }
}

// ---------------------------------------------------------------------------
// Causal flash attention, one wave (32 threads) per 16-query tile per (b,h).
// Q,K: bf16 [bh][S][64]; Vt: bf16 [bh][64][S]; out Z: bf16 [b*S+s][h*64+dk].
// Per 32-key chunk: 4 wmma for scores, online softmax with shfl row
// reductions, P -> LDS -> A fragment, 4 wmma for P.V.  Q tile is staged
// into LDS by the TDM once per block.
// ---------------------------------------------------------------------------
__global__ void __launch_bounds__(32)
attn_fwd(const bf16_t* __restrict__ Qb, const bf16_t* __restrict__ Kb,
         const bf16_t* __restrict__ Vt, bf16_t* __restrict__ Zb) {
    constexpr int LP = 40;            // 16x32 P tile, padded row stride
    __shared__ bf16_t Pt[16 * LP];
    const int lane = threadIdx.x;
    const int s0   = blockIdx.x * 16;
    const int bh   = blockIdx.y;
    const int bIdx = bh >> 4, hIdx = bh & 15;

    const bf16_t* Q  = Qb + (size_t)bh * Sc * DKc + (size_t)s0 * DKc;
    const bf16_t* Kp = Kb + (size_t)bh * Sc * DKc;
    const bf16_t* Vp = Vt + (size_t)bh * DKc * Sc;

    const int nn    = lane & 15;
    const int mhalf = (lane >> 4) << 3;

    // Q fragments live in registers for the whole tile (2 x K=32 chunks)
#if HAVE_TDM
    __shared__ bf16_t Qs[16 * DKc];   // 2KB, contiguous copy of the Q tile
    tdm_load_2d_bf16((unsigned)(uintptr_t)&Qs[0], Q,
                     /*tile_d0=*/16 * DKc, /*tile_d1=*/1,
                     /*stride0=*/(unsigned long long)(16 * DKc),
                     0u, 0u, false);
    __builtin_amdgcn_s_wait_tensorcnt(0);
    const v16bf qf0 = load_fragA(Qs, DKc, lane);
    const v16bf qf1 = load_fragA(Qs + 32, DKc, lane);
#else
    const v16bf qf0 = load_fragA(Q, DKc, lane);
    const v16bf qf1 = load_fragA(Q + 32, DKc, lane);
#endif

    float mst[8], lst[8];
#pragma unroll
    for (int i = 0; i < 8; ++i) { mst[i] = -3.0e38f; lst[i] = 0.0f; }
    v8f o0 = {}, o1 = {}, o2 = {}, o3 = {};

    const int nch = (s0 + 16 + 31) >> 5;          // 32-key chunks, causal
    for (int tc = 0; tc < nch; ++tc) {
        const int t0 = tc << 5;
        if (tc + 1 < nch) {  // warm L2/WGP$ for the next chunk
            __builtin_prefetch(Kp + (size_t)(t0 + 32 + (lane >> 1)) * DKc, 0, 0);
            __builtin_prefetch(Vp + (size_t)lane * Sc + t0 + 32, 0, 0);
            __builtin_prefetch(Vp + (size_t)(32 + lane) * Sc + t0 + 32, 0, 0);
        }
        // ---- scores: S[m][n] = Q[m,:] . K[t0+n,:]  (two 16-key halves) ----
        v8f sc0 = {}, sc1 = {};
        {
            const bf16_t* K0 = Kp + (size_t)t0 * DKc;
            sc0 = wmma_bf16(qf0, load_fragB(K0, DKc, lane), sc0);
            sc0 = wmma_bf16(qf1, load_fragB(K0 + 32, DKc, lane), sc0);
            const bf16_t* K1 = Kp + (size_t)(t0 + 16) * DKc;
            sc1 = wmma_bf16(qf0, load_fragB(K1, DKc, lane), sc1);
            sc1 = wmma_bf16(qf1, load_fragB(K1 + 32, DKc, lane), sc1);
        }
        // ---- scale, causal mask, online softmax, emit P into LDS ----
#pragma unroll
        for (int vg = 0; vg < 8; ++vg) {
            const int qrow = s0 + vg + mhalf;
            float x0 = sc0[vg] * 0.125f;          // 1/sqrt(64)
            float x1 = sc1[vg] * 0.125f;
            if (t0 + nn      > qrow) x0 = -3.0e38f;
            if (t0 + 16 + nn > qrow) x1 = -3.0e38f;
            float rm = fmaxf(x0, x1);
#pragma unroll
            for (int off = 1; off < 16; off <<= 1)   // row max: xor masks 1..8
                rm = fmaxf(rm, __shfl_xor(rm, off, 32));
            const float mnew  = fmaxf(mst[vg], rm);
            const float alpha = __expf(mst[vg] - mnew);
            const float p0    = __expf(x0 - mnew);
            const float p1    = __expf(x1 - mnew);
            float rs = p0 + p1;
#pragma unroll
            for (int off = 1; off < 16; off <<= 1)   // row sum
                rs += __shfl_xor(rs, off, 32);
            lst[vg] = lst[vg] * alpha + rs;
            mst[vg] = mnew;
            o0[vg] *= alpha; o1[vg] *= alpha; o2[vg] *= alpha; o3[vg] *= alpha;
            const int prow = vg + mhalf;             // C-layout -> row-major P
            Pt[prow * LP + nn]      = f2bf(p0);
            Pt[prow * LP + 16 + nn] = f2bf(p1);
        }
        __syncthreads();
        // ---- O += P . V : A = P (16x32), B = Vt rows (dk-major) ----
        const v16bf pf = load_fragA(Pt, LP, lane);
        o0 = wmma_bf16(pf, load_fragB(Vp + (size_t) 0 * Sc + t0, Sc, lane), o0);
        o1 = wmma_bf16(pf, load_fragB(Vp + (size_t)16 * Sc + t0, Sc, lane), o1);
        o2 = wmma_bf16(pf, load_fragB(Vp + (size_t)32 * Sc + t0, Sc, lane), o2);
        o3 = wmma_bf16(pf, load_fragB(Vp + (size_t)48 * Sc + t0, Sc, lane), o3);
        __syncthreads();
    }
    // ---- normalize and store Z with heads concatenated on the feature dim --
#pragma unroll
    for (int vg = 0; vg < 8; ++vg) {
        const float inv = 1.0f / lst[vg];
        const int srow  = s0 + vg + mhalf;
        bf16_t* Z = Zb + ((size_t)bIdx * Sc + srow) * Dc + hIdx * DKc + nn;
        Z[0]  = f2bf(o0[vg] * inv);
        Z[16] = f2bf(o1[vg] * inv);
        Z[32] = f2bf(o2[vg] * inv);
        Z[48] = f2bf(o3[vg] * inv);
    }
}

// ---------------------------------------------------------------------------
// Host-side launcher. Workspace: Qbf | Kbf | Vt(bf,transposed) | Zbf,
// 16 MiB each (64 MiB total) -> the whole attention working set sits in L2.
// ---------------------------------------------------------------------------
extern "C" void kernel_launch(void* const* d_in, const int* in_sizes, int n_in,
                              void* d_out, int out_size, void* d_ws,
                              size_t ws_size, hipStream_t stream) {
    (void)in_sizes; (void)n_in; (void)out_size; (void)ws_size;
    const float* query = (const float*)d_in[0];
    const float* key   = (const float*)d_in[1];
    const float* value = (const float*)d_in[2];
    const float* Wq = (const float*)d_in[3];
    const float* bq = (const float*)d_in[4];
    const float* Wk = (const float*)d_in[5];
    const float* bk = (const float*)d_in[6];
    const float* Wv = (const float*)d_in[7];
    const float* bv = (const float*)d_in[8];
    const float* Wp = (const float*)d_in[9];
    const float* bp = (const float*)d_in[10];
    // d_in[11] (mask) is implied by the causal loop structure.
    float* out = (float*)d_out;

    constexpr size_t QKV_BYTES = (size_t)Bc * Hc * Sc * DKc * sizeof(bf16_t);
    char* ws = (char*)d_ws;
    bf16_t* Qw = (bf16_t*)(ws);
    bf16_t* Kw = (bf16_t*)(ws + QKV_BYTES);
    bf16_t* Vw = (bf16_t*)(ws + 2 * QKV_BYTES);
    bf16_t* Zw = (bf16_t*)(ws + 3 * QKV_BYTES);

    const dim3 gg(Mrows / BM, Dc / BN);   // 128 x 32 blocks
    const dim3 gb(256);

    gemm_wmma_bias<0, false><<<gg, gb, 0, stream>>>(query, Wq, bq, Qw,
                                                    Mrows, Dc, Dc);
    gemm_wmma_bias<0, false><<<gg, gb, 0, stream>>>(key, Wk, bk, Kw,
                                                    Mrows, Dc, Dc);
    gemm_wmma_bias<1, false><<<gg, gb, 0, stream>>>(value, Wv, bv, Vw,
                                                    Mrows, Dc, Dc);

    attn_fwd<<<dim3(Sc / 16, Bc * Hc), dim3(32), 0, stream>>>(Qw, Kw, Vw, Zw);

    gemm_wmma_bias<2, true><<<gg, gb, 0, stream>>>(Zw, Wp, bp, out,
                                                   Mrows, Dc, Dc);
}
